// LSTM_2233382994362
// MI455X (gfx1250) — matrix operand order repaired and checked by reading, not compiled, not run on hardware
//
#include <hip/hip_runtime.h>
#include <hip/hip_bf16.h>

typedef _Float16 h8   __attribute__((ext_vector_type(8)));
typedef _Float16 v16h __attribute__((ext_vector_type(16)));
typedef float    v8f  __attribute__((ext_vector_type(8)));

#define LSTM_B 32
#define LSTM_T 512
#define LSTM_H 1024
#define LSTM_BH (LSTM_B * LSTM_H)      // 32768
#define LSTM_BTH ((long)LSTM_B * LSTM_T * LSTM_H)  // 16,777,216

// ---- fragment loads (per CDNA5 ISA 7.12.2 lane layouts) ----
// A 16x32 f16: lanes 0-15 hold row m=lane, K = {0..7, 16..23}; lanes 16-31: K = {8..15, 24..31}
__device__ __forceinline__ v16h ld_a(const _Float16* __restrict__ row, int k0, int sel) {
    h8 lo = *(const h8*)(row + k0 + sel * 8);
    h8 hi = *(const h8*)(row + k0 + 16 + sel * 8);
    return __builtin_shufflevector(lo, hi, 0,1,2,3,4,5,6,7,8,9,10,11,12,13,14,15);
}
// B 32x16 f16: lanes 0-15 hold col n=lane, K = {0..7, 16..23}; lanes 16-31: K = {8..15, 24..31}
__device__ __forceinline__ v16h ld_b(const _Float16* __restrict__ row, int k0, int sel) {
    h8 lo = *(const h8*)(row + k0 + sel * 8);
    h8 hi = *(const h8*)(row + k0 + 16 + sel * 8);
    return __builtin_shufflevector(lo, hi, 0,1,2,3,4,5,6,7,8,9,10,11,12,13,14,15);
}

__device__ __forceinline__ float sigm_f(float x) { return 1.0f / (1.0f + __expf(-x)); }
__device__ __forceinline__ float tanh_f(float x) {
    float a = fabsf(x);
    float e = __expf(-2.0f * a);
    float t = (1.0f - e) / (1.0f + e);
    return x < 0.0f ? -t : t;
}

// sense-reversing grid barrier (32 co-resident WGs)
__device__ __forceinline__ void grid_barrier(unsigned* cnt, unsigned* gen, unsigned nb) {
    __syncthreads();
    if (threadIdx.x == 0) {
        __threadfence();
        unsigned old  = __hip_atomic_load(gen, __ATOMIC_RELAXED, __HIP_MEMORY_SCOPE_AGENT);
        unsigned prev = __hip_atomic_fetch_add(cnt, 1u, __ATOMIC_ACQ_REL, __HIP_MEMORY_SCOPE_AGENT);
        if (prev == nb - 1u) {
            __hip_atomic_store(cnt, 0u, __ATOMIC_RELAXED, __HIP_MEMORY_SCOPE_AGENT);
            __hip_atomic_fetch_add(gen, 1u, __ATOMIC_ACQ_REL, __HIP_MEMORY_SCOPE_AGENT);
        } else {
            while (__hip_atomic_load(gen, __ATOMIC_ACQUIRE, __HIP_MEMORY_SCOPE_AGENT) == old)
                __builtin_amdgcn_s_sleep(1);
        }
    }
    __syncthreads();
}

// Persistent per-layer LSTM kernel. Grid = 32 WGs x 256 threads (8 wave32).
// WG bx owns hidden slice j0..j0+31 across all 4 gates. Wave w: gate = w>>1,
// 16-column strip (w&1)*16; computes 32(M) x 16(N) over K=1024(x) + 1024(h).
__global__ __launch_bounds__(256) void lstm_layer_persistent(
    const _Float16* __restrict__ x, long xT, long xRow,
    const _Float16* __restrict__ Wih, const _Float16* __restrict__ Whh,
    const float* __restrict__ bih, const float* __restrict__ bhh,
    const float* __restrict__ c0,
    _Float16* __restrict__ hBuf, int hMod,
    float* __restrict__ out, long outRow,
    float* __restrict__ hFin, float* __restrict__ cFin,
    unsigned* barCnt, unsigned* barGen, int nsteps)
{
    const int tid  = threadIdx.x;
    const int lane = tid & 31;
    const int wave = tid >> 5;
    const int sel  = lane >> 4;
    const int j0   = blockIdx.x * 32;

    // matmul role
    const int gate = wave >> 1;
    const int jW   = ((wave & 1) << 4) + (lane & 15);   // 0..31 within slice
    const int gcol = gate * LSTM_H + j0 + jW;           // row of W  ([4H,H] row-major)
    const _Float16* wihRow = Wih + (long)gcol * LSTM_H;
    const _Float16* whhRow = Whh + (long)gcol * LSTM_H;
    const float bias = bih[gcol] + bhh[gcol];
    const int mA0 = lane & 15;
    const int mA1 = mA0 + 16;

    // elementwise role: thread -> (j = tid&31, m = (tid>>5)+k*8), coalesced in j
    const int jj = tid & 31;
    const int mb = tid >> 5;
    float cReg[4], hLast[4];
#pragma unroll
    for (int k = 0; k < 4; ++k) {
        int m = mb + k * 8;
        cReg[k]  = c0[m * LSTM_H + j0 + jj];
        hLast[k] = 0.0f;
    }

    __shared__ float lds[4][32][33];   // [gate][j][m], padded

    for (int t = 0; t < nsteps; ++t) {
        const _Float16* xA = x + (long)t * xT;
        const _Float16* hA = hBuf + (long)(t % hMod) * LSTM_BH;
        const _Float16* xr0 = xA + (long)mA0 * xRow;
        const _Float16* xr1 = xA + (long)mA1 * xRow;
        const _Float16* hr0 = hA + mA0 * LSTM_H;
        const _Float16* hr1 = hA + mA1 * LSTM_H;

        v8f acc0 = {}, acc1 = {};
#pragma unroll 4
        for (int k0 = 0; k0 < LSTM_H; k0 += 32) {       // input projection
            v16h b  = ld_b(wihRow, k0, sel);
            v16h a0 = ld_a(xr0, k0, sel);
            v16h a1 = ld_a(xr1, k0, sel);
            acc0 = __builtin_amdgcn_wmma_f32_16x16x32_f16(false, a0, false, b, (short)0, acc0, false, false);
            acc1 = __builtin_amdgcn_wmma_f32_16x16x32_f16(false, a1, false, b, (short)0, acc1, false, false);
        }
#pragma unroll 4
        for (int k0 = 0; k0 < LSTM_H; k0 += 32) {       // recurrent projection
            v16h b  = ld_b(whhRow, k0, sel);
            v16h a0 = ld_a(hr0, k0, sel);
            v16h a1 = ld_a(hr1, k0, sel);
            acc0 = __builtin_amdgcn_wmma_f32_16x16x32_f16(false, a0, false, b, (short)0, acc0, false, false);
            acc1 = __builtin_amdgcn_wmma_f32_16x16x32_f16(false, a1, false, b, (short)0, acc1, false, false);
        }

        // C/D layout: acc[r] at (m = r + 8*sel [+16 for acc1], n = lane&15)
#pragma unroll
        for (int r = 0; r < 8; ++r) {
            lds[gate][jW][(sel << 3) + r]      = acc0[r] + bias;
            lds[gate][jW][16 + (sel << 3) + r] = acc1[r] + bias;
        }
        __syncthreads();

        _Float16* hW = hBuf + (long)((t + 1) % hMod) * LSTM_BH;
#pragma unroll
        for (int k = 0; k < 4; ++k) {
            int m = mb + k * 8;
            float iv = lds[0][jj][m];
            float fv = lds[1][jj][m];
            float gv = lds[2][jj][m];
            float ov = lds[3][jj][m];
            float cn = sigm_f(fv) * cReg[k] + sigm_f(iv) * tanh_f(gv);
            float hv = sigm_f(ov) * tanh_f(cn);
            cReg[k]  = cn;
            hLast[k] = hv;
            hW[m * LSTM_H + j0 + jj] = (_Float16)hv;
            if (out) out[(long)m * outRow + (long)t * LSTM_H + j0 + jj] = hv;
        }
        grid_barrier(barCnt, barGen, gridDim.x);
    }

#pragma unroll
    for (int k = 0; k < 4; ++k) {
        int m = mb + k * 8;
        hFin[m * LSTM_H + j0 + jj] = hLast[k];
        cFin[m * LSTM_H + j0 + jj] = cReg[k];
    }
}

__global__ void cvt_f32_f16(const float* __restrict__ s, _Float16* __restrict__ d, long n) {
    long i = (long)blockIdx.x * blockDim.x + threadIdx.x;
    long stride = (long)gridDim.x * blockDim.x;
    for (; i < n; i += stride) d[i] = (_Float16)s[i];
}

__global__ void zero_u32(unsigned* p, int n) {
    int i = threadIdx.x;
    if (i < n) p[i] = 0u;
}

extern "C" void kernel_launch(void* const* d_in, const int* in_sizes, int n_in,
                              void* d_out, int out_size, void* d_ws, size_t ws_size,
                              hipStream_t stream) {
    (void)in_sizes; (void)n_in; (void)out_size; (void)ws_size;
    const float* in_x = (const float*)d_in[0];
    const float* h0   = (const float*)d_in[1];
    const float* c0   = (const float*)d_in[2];
    const float* Wih  = (const float*)d_in[3];
    const float* Whh  = (const float*)d_in[4];
    const float* bih  = (const float*)d_in[5];
    const float* bhh  = (const float*)d_in[6];
    float* out = (float*)d_out;

    const long NW = 2L * 4 * LSTM_H * LSTM_H;  // elements per weight tensor (both layers)
    char* ws = (char*)d_ws;
    _Float16* xf16  = (_Float16*)(ws);                                   // B*T*H
    _Float16* Wih16 = (_Float16*)(ws + 33554432);                        // 8,388,608 halfs
    _Float16* Whh16 = (_Float16*)(ws + 50331648);
    _Float16* hs1   = (_Float16*)(ws + 67108864);                        // [T+1, B, H]
    _Float16* hping = (_Float16*)(ws + 100728832);                       // [2, B, H]
    unsigned* bar   = (unsigned*)(ws + 100859904);                       // {cnt, gen}

    // precision conversions (one pass each, fully parallel)
    cvt_f32_f16<<<4096, 256, 0, stream>>>(in_x, xf16, LSTM_BTH);
    cvt_f32_f16<<<4096, 256, 0, stream>>>(Wih, Wih16, NW);
    cvt_f32_f16<<<4096, 256, 0, stream>>>(Whh, Whh16, NW);
    cvt_f32_f16<<<128, 256, 0, stream>>>(h0, hs1, LSTM_BH);              // layer0 h init -> row 0
    cvt_f32_f16<<<128, 256, 0, stream>>>(h0 + LSTM_BH, hping, LSTM_BH);  // layer1 h init -> ping 0
    zero_u32<<<1, 64, 0, stream>>>(bar, 2);

    // layer 0: x = input f16 ([B,T,H]); h history in hs1 rows 1..T
    lstm_layer_persistent<<<32, 256, 0, stream>>>(
        xf16, (long)LSTM_H, (long)LSTM_T * LSTM_H,
        Wih16, Whh16, bih, bhh, c0,
        hs1, LSTM_T + 1,
        nullptr, 0,
        out + LSTM_BTH,                       // h_n[0]
        out + LSTM_BTH + 2 * LSTM_BH,         // c_n[0]
        bar, bar + 1, LSTM_T);

    // layer 1: x = layer-0 h history (hs1 rows 1..T, [T,B,H]); h ping-pong
    lstm_layer_persistent<<<32, 256, 0, stream>>>(
        hs1 + LSTM_BH, (long)LSTM_BH, (long)LSTM_H,
        Wih16 + NW / 2, Whh16 + NW / 2, bih + 4 * LSTM_H, bhh + 4 * LSTM_H, c0 + LSTM_BH,
        hping, 2,
        out, (long)LSTM_T * LSTM_H,           // output [B,T,H]
        out + LSTM_BTH + LSTM_BH,             // h_n[1]
        out + LSTM_BTH + 3 * LSTM_BH,         // c_n[1]
        bar, bar + 1, LSTM_T);
}
